// LeafBlockAttention_36627481101009
// MI455X (gfx1250) — matrix operand order repaired and checked by reading, not compiled, not run on hardware
//
#include <hip/hip_runtime.h>
#include <hip/hip_bf16.h>

// ---------------------------------------------------------------------------
// LeafBlockAttention for MI455X (gfx1250, wave32, WMMA, TDM)
//
// Per-block fused pipeline (1 workgroup = 1 leaf block, 256 threads = 8 waves):
//   TDM tensor_load_to_lds: x block (32x512 f32) -> LDS (ASYNC, wave 0)
//   LDS convert -> X bf16 (+mean row)
//   QKV = X(48x512,pad) @ WqkvT(1536x512 bf16, streamed from L2)   [WMMA bf16]
//   S   = Q @ K^T per head (32x48 pad), +bias, mask -> -inf        [WMMA bf16]
//   softmax + edge-gate term -> combined (32x64 pad, bf16)
//   O   = combined @ V per head                                    [WMMA bf16]
//   out = O(32x512) @ WpT(512x512 bf16) + proj_b  -> f32 global    [WMMA bf16]
// ---------------------------------------------------------------------------

typedef __bf16 bf16;
typedef __attribute__((ext_vector_type(16))) __bf16 v16bf;
typedef __attribute__((ext_vector_type(8)))  __bf16 v8bf;
typedef __attribute__((ext_vector_type(8)))  float  v8f;
typedef __attribute__((ext_vector_type(4)))  unsigned int u32x4;
typedef __attribute__((ext_vector_type(8)))  int i32x8;
typedef __attribute__((ext_vector_type(4)))  int i32x4;

#define LEAF    32
#define NBLK    4096
#define NNODES  131072
#define DIM     512
#define NH      8
#define HD      64
#define QKV_N   1536
#define NEDGES  2097152

// ---- LDS layout (bytes) ---------------------------------------------------
// o0: Xb  bf16[48][512]  = 49152   (reused as Sf f32[8][32][48] = 49152)
// o1: Qb  bf16[32][512]  = 32768   (reused as Ab bf16[32][512])
// o2: Kb  bf16[48][512]  = 49152
// o3: Vt  bf16[512][64]  = 65536   (staging: Xf f32[32][512] via TDM)
// o4: Cb  bf16[8][32][64]= 32768
// o5: biasL f32[32][33]  = 4224
// o6: maskL u8 [32][33]  = 1056
#define O0 0
#define O1 49152
#define O2 (49152 + 32768)
#define O3 (49152 + 32768 + 49152)
#define O4 (49152 + 32768 + 49152 + 65536)
#define O5 (49152 + 32768 + 49152 + 65536 + 32768)
#define O6 (O5 + 4224)
#define LDS_BYTES (O6 + 1056)

#if defined(__has_builtin)
#if __has_builtin(__builtin_amdgcn_tensor_load_to_lds) && \
    __has_builtin(__builtin_amdgcn_s_wait_tensorcnt)
#define USE_TDM 1
#endif
#endif

// ---------------------------------------------------------------------------
// Fragment loaders for V_WMMA_F32_16X16X32_BF16 (wave32; this code's contiguous k order, same in A and B)
// A 16x32: lanes 0-15 hold M=lane, K = k0..k0+15 contiguous;
//          lanes 16-31 hold M=lane-16, K = k0+16..k0+31 contiguous.
// B 32x16: lanes 0-15 hold N=lane, K = k0..k0+15 contiguous;
//          lanes 16-31 hold N=lane-16, K = k0+16..k0+31 contiguous.
//          (source matrices stored [N][K] so B loads are contiguous)
// C/D 16x16 f32: VGPR j holds M = j + (lane<16 ? 0 : 8), N = lane%16.
// ---------------------------------------------------------------------------
__device__ __forceinline__ v16bf load_a_frag(const bf16* base, int ld,
                                             int m0, int k0, int lane) {
  int m = m0 + (lane & 15);
  int kb = k0 + ((lane >> 4) << 4);        // +16 elements for upper half-wave
  union { v16bf v; v8bf h[2]; } u;
  u.h[0] = *(const v8bf*)(base + m * ld + kb);
  u.h[1] = *(const v8bf*)(base + m * ld + kb + 8);
  return u.v;
}

__device__ __forceinline__ v16bf load_b_frag(const bf16* base, int ld,
                                             int n0, int k0, int lane) {
  int n  = n0 + (lane & 15);
  int kb = k0 + ((lane >> 4) << 4);        // +16 elements for upper half-wave
  union { v16bf v; v8bf h[2]; } u;
  u.h[0] = *(const v8bf*)(base + n * ld + kb);
  u.h[1] = *(const v8bf*)(base + n * ld + kb + 8);
  return u.v;
}

__device__ __forceinline__ v8f wmma_bf16(v16bf a, v16bf b, v8f c) {
  return __builtin_amdgcn_wmma_f32_16x16x32_bf16(
      /*neg_a=*/false, a, /*neg_b=*/false, b,
      /*c_mod=*/(short)0, c, /*reuse_a=*/false, /*reuse_b=*/false);
}

// ---------------------------------------------------------------------------
// Pre-kernels
// ---------------------------------------------------------------------------
__global__ void convert_weights_kernel(const float* __restrict__ qkv_w,
                                       const float* __restrict__ proj_w,
                                       bf16* __restrict__ wq_t,
                                       bf16* __restrict__ wp_t) {
  int i = blockIdx.x * blockDim.x + threadIdx.x;
  if (i < DIM * QKV_N) {                       // qkv_w [512][1536] -> [1536][512]
    int k = i / QKV_N, n = i % QKV_N;
    wq_t[n * DIM + k] = (bf16)qkv_w[i];
  }
  int j = i - DIM * QKV_N;
  if (j >= 0 && j < DIM * DIM) {               // proj_w [512][512] -> [512][512]^T
    int k = j / DIM, n = j % DIM;
    wp_t[n * DIM + k] = (bf16)proj_w[j];
  }
}

__global__ void init_mask_bias_kernel(float* __restrict__ bias,
                                      unsigned char* __restrict__ mask) {
  long i = (long)blockIdx.x * blockDim.x + threadIdx.x;
  if (i >= (long)NBLK * LEAF * (LEAF + 1)) return;
  int c = (int)(i % (LEAF + 1));
  int r = (int)((i / (LEAF + 1)) % LEAF);
  bool on = (c == LEAF) || (c == r);
  bias[i] = on ? 1.0f : 0.0f;
  mask[i] = on ? (unsigned char)1 : (unsigned char)0;
}

__global__ void scatter_edges_kernel(const int* __restrict__ rows,
                                     const int* __restrict__ cols,
                                     const float* __restrict__ vals,
                                     float* __restrict__ bias,
                                     unsigned char* __restrict__ mask) {
  int e = blockIdx.x * blockDim.x + threadIdx.x;
  if (e >= NEDGES) return;
  int r = rows[e], c = cols[e];
  int br = r / LEAF, bc = c / LEAF;
  if (br != bc || br >= NBLK) return;          // mode='drop'
  int rl = r % LEAF, cl = c % LEAF;
  long idx = ((long)br * LEAF + rl) * (LEAF + 1) + cl;
  mask[idx] = 1;
  if (rl != cl) bias[idx] = vals[e];           // diagonal overridden to 1 by ref
}

// ---------------------------------------------------------------------------
// Fused per-block kernel
// ---------------------------------------------------------------------------
__global__ void __launch_bounds__(256)
leaf_attn_kernel(const float* __restrict__ x,
                 const float* __restrict__ qkv_b,
                 const float* __restrict__ proj_b,
                 const float* __restrict__ eg_w,
                 const float* __restrict__ eg_b,
                 const bf16*  __restrict__ wq_t,
                 const bf16*  __restrict__ wp_t,
                 const float* __restrict__ bias_g,
                 const unsigned char* __restrict__ mask_g,
                 float* __restrict__ out) {
  extern __shared__ char smem[];
  bf16*  Xb    = (bf16*)(smem + O0);           // [48][512]
  float* Sf    = (float*)(smem + O0);          // [8][32][48] (reuses Xb)
  bf16*  Qb    = (bf16*)(smem + O1);           // [32][512]
  bf16*  Ab    = (bf16*)(smem + O1);           // [32][512] (reuses Qb)
  bf16*  Kb    = (bf16*)(smem + O2);           // [48][512]
  bf16*  Vt    = (bf16*)(smem + O3);           // [512][64]  (V transposed)
  float* Xf    = (float*)(smem + O3);          // [32][512] f32 staging (pre-Vt)
  bf16*  Cb    = (bf16*)(smem + O4);           // [8][32][64]
  float* biasL = (float*)(smem + O5);          // [32][33]
  unsigned char* maskL = (unsigned char*)(smem + O6);

  const int tid  = threadIdx.x;
  const int lane = tid & 31;
  const int wave = tid >> 5;
  const int colN = lane & 15;
  const int blk  = blockIdx.x;

  const float* xblk = x + (long)blk * LEAF * DIM;

  // ---- Phase 0a: stage x block (32x512 f32 = 64KB) into LDS ----
#if USE_TDM
  if (wave == 0) {
    // Tensor DMA descriptor (CDNA5 ISA):
    //  2D tile 512x32 of 4-byte elements, contiguous (stride 512).
    unsigned long long ga = (unsigned long long)(const void*)xblk;
    u32x4 g0;
    g0.x = 1u;                                           // count=1, user mode
    g0.y = (unsigned int)O3;                             // lds_addr
    g0.z = (unsigned int)(ga & 0xFFFFFFFFu);             // global_addr[31:0]
    g0.w = (unsigned int)((ga >> 32) & 0x01FFFFFFu)      // global_addr[56:32]
         | (2u << 30);                                   // type = 2 ("image")
    i32x8 g1;
    g1[0] = (2 << 16);                    // data_size=2 (4B), wg_mask=0, flags=0
    g1[1] = (DIM & 0xFFFF) << 16;         // tensor_dim0[15:0]=512 (abar=0)
    g1[2] = (DIM >> 16) | (LEAF << 16);   // tensor_dim0 hi | tensor_dim1=32
    g1[3] = (DIM << 16);                  // tensor_dim1 hi=0 | tile_dim0=512
    g1[4] = LEAF;                         // tile_dim1=32, tile_dim2=0
    g1[5] = DIM;                          // tensor_dim0_stride=512 (lo32)
    g1[6] = 0;                            // stride0 hi | stride1 lo
    g1[7] = 0;                            // stride1 hi
    i32x4 gz4 = {0, 0, 0, 0};             // groups 2/3 unused (<=2D tensor)
    i32x8 gz8 = {0, 0, 0, 0, 0, 0, 0, 0}; // extra SGPR group (6-arg variant)
    __builtin_amdgcn_tensor_load_to_lds(g0, g1, gz4, gz4, gz8, 0);
    __builtin_amdgcn_s_wait_tensorcnt((short)0);
  }
  __syncthreads();
#else
  for (int i = tid; i < LEAF * DIM; i += 256) Xf[i] = xblk[i];
  __syncthreads();
#endif

  // ---- Phase 0b: convert to bf16, mean row, pads, mask/bias (from LDS) ----
  for (int i = tid; i < LEAF * DIM; i += 256)
    Xb[i] = (bf16)Xf[i];
  for (int c = tid; c < DIM; c += 256) {
    float s = 0.f;
    for (int r = 0; r < LEAF; ++r) s += Xf[r * DIM + c];
    Xb[32 * DIM + c] = (bf16)(s * (1.0f / LEAF));
    for (int r = 33; r < 48; ++r) { Xb[r * DIM + c] = (bf16)0.f; Kb[r * DIM + c] = (bf16)0.f; }
  }
  for (int i = tid; i < LEAF * (LEAF + 1); i += 256) {
    long g = (long)blk * LEAF * (LEAF + 1) + i;
    biasL[i] = bias_g[g];
    maskL[i] = mask_g[g];
  }
  __syncthreads();   // Xf fully consumed; its region becomes Vt below

  // zero V^T pad keys 33..63 (disjoint from Vt rows written in phase 1)
  for (int i = tid; i < DIM * (64 - 33); i += 256) {
    int d = i / 31, kk = 33 + (i % 31);
    Vt[d * 64 + kk] = (bf16)0.f;
  }

  // ---- Phase 1: QKV GEMM  [48x512] @ [512x1536] (B streamed from L2) ----
  for (int nt = wave; nt < QKV_N / 16; nt += 8) {
    if (nt + 8 < QKV_N / 16)   // prefetch next B panel (global_prefetch_b8)
      __builtin_prefetch(wq_t + ((nt + 8) * 16 + colN) * DIM, 0, 0);
    int n_abs = nt * 16 + colN;
    float bq = qkv_b[n_abs];
    for (int mt = 0; mt < 3; ++mt) {
      v8f acc = {};
      for (int ks = 0; ks < DIM / 32; ++ks) {
        v16bf a = load_a_frag(Xb, DIM, mt * 16, ks * 32, lane);
        v16bf b = load_b_frag(wq_t, DIM, nt * 16, ks * 32, lane);
        acc = wmma_bf16(a, b, acc);
      }
      int rbase = mt * 16 + ((lane >> 4) << 3);
      for (int j = 0; j < 8; ++j) {
        int r = rbase + j;
        float val = acc[j] + bq;
        if (n_abs < DIM)            { if (r < 32) Qb[r * DIM + n_abs] = (bf16)val; }
        else if (n_abs < 2 * DIM)   { if (r < 33) Kb[r * DIM + (n_abs - DIM)] = (bf16)val; }
        else                        { if (r < 33) Vt[(n_abs - 2 * DIM) * 64 + r] = (bf16)val; }
      }
    }
  }
  __syncthreads();

  // ---- Phase 2: scores S = Q @ K^T per head, +bias, mask ----
  for (int t = wave; t < 48; t += 8) {
    int h = t / 6, rem = t % 6, mt = rem / 3, nt = rem % 3;
    v8f acc = {};
    for (int ks = 0; ks < 2; ++ks) {
      v16bf a = load_a_frag(Qb, DIM, mt * 16, h * HD + ks * 32, lane);
      v16bf b = load_b_frag(Kb, DIM, nt * 16, h * HD + ks * 32, lane);
      acc = wmma_bf16(a, b, acc);
    }
    int key = nt * 16 + colN;
    int rbase = mt * 16 + ((lane >> 4) << 3);
    for (int j = 0; j < 8; ++j) {
      int q = rbase + j;
      float sval = -__builtin_inff();
      if (key <= LEAF && maskL[q * (LEAF + 1) + key])
        sval = acc[j] * 0.125f + biasL[q * (LEAF + 1) + key];   // scale = 64^-0.5
      Sf[(h * LEAF + q) * 48 + key] = sval;
    }
  }
  __syncthreads();

  // ---- Phase 3: softmax over 33 keys + edge-gate term -> combined (bf16) ----
  {
    int h = tid >> 5, q = tid & 31;
    float mx = -__builtin_inff();
    for (int k = 0; k <= LEAF; ++k)
      mx = fmaxf(mx, Sf[(h * LEAF + q) * 48 + k]);
    float ssum = 0.f;
    for (int k = 0; k <= LEAF; ++k)
      ssum += __expf(Sf[(h * LEAF + q) * 48 + k] - mx);
    float inv = 1.0f / ssum;
    float ew = eg_w[h], eb = eg_b[h];
    for (int k = 0; k < 64; ++k) {
      float v = 0.f;
      if (k <= LEAF) {
        int mi = q * (LEAF + 1) + k;
        if (maskL[mi])
          v = __expf(Sf[(h * LEAF + q) * 48 + k] - mx) * inv + (biasL[mi] * ew + eb);
      }
      Cb[(h * LEAF + q) * 64 + k] = (bf16)v;
    }
  }
  __syncthreads();

  // ---- Phase 4: O = combined @ V per head ----
  for (int t = wave; t < 64; t += 8) {
    int h = t >> 3, rem = t & 7, mt = rem >> 2, nt = rem & 3;
    v8f acc = {};
    for (int ks = 0; ks < 2; ++ks) {
      v16bf a = load_a_frag(Cb + h * LEAF * 64, 64, mt * 16, ks * 32, lane);
      v16bf b = load_b_frag(Vt, 64, h * HD + nt * 16, ks * 32, lane);
      acc = wmma_bf16(a, b, acc);
    }
    int d = h * HD + nt * 16 + colN;
    int rbase = mt * 16 + ((lane >> 4) << 3);
    for (int j = 0; j < 8; ++j)
      Ab[(rbase + j) * DIM + d] = (bf16)acc[j];
  }
  __syncthreads();

  // ---- Phase 5: out = O @ Wp + proj_b (f32 to global) ----
  float* outblk = out + (long)blk * LEAF * DIM;
  for (int nt = wave; nt < DIM / 16; nt += 8) {
    if (nt + 8 < DIM / 16)
      __builtin_prefetch(wp_t + ((nt + 8) * 16 + colN) * DIM, 0, 0);
    int n_abs = nt * 16 + colN;
    float bp = proj_b[n_abs];
    for (int mt = 0; mt < 2; ++mt) {
      v8f acc = {};
      for (int ks = 0; ks < DIM / 32; ++ks) {
        v16bf a = load_a_frag(Ab, DIM, mt * 16, ks * 32, lane);
        v16bf b = load_b_frag(wp_t, DIM, nt * 16, ks * 32, lane);
        acc = wmma_bf16(a, b, acc);
      }
      int rbase = mt * 16 + ((lane >> 4) << 3);
      for (int j = 0; j < 8; ++j)
        outblk[(rbase + j) * DIM + n_abs] = acc[j] + bp;
    }
  }
}

// ---------------------------------------------------------------------------
extern "C" void kernel_launch(void* const* d_in, const int* in_sizes, int n_in,
                              void* d_out, int out_size, void* d_ws, size_t ws_size,
                              hipStream_t stream) {
  const float* x      = (const float*)d_in[0];
  const int*   ei     = (const int*)  d_in[1];   // [2][NEDGES]
  const float* ev     = (const float*)d_in[2];
  // d_in[3] positions: unused by reference math
  const float* qkv_w  = (const float*)d_in[4];
  const float* qkv_b  = (const float*)d_in[5];
  const float* proj_w = (const float*)d_in[6];
  const float* proj_b = (const float*)d_in[7];
  const float* eg_w   = (const float*)d_in[8];
  const float* eg_b   = (const float*)d_in[9];
  float* out = (float*)d_out;

  char* ws = (char*)d_ws;
  const size_t wq_bytes   = (size_t)QKV_N * DIM * sizeof(bf16);      // 1.5 MB
  const size_t wp_bytes   = (size_t)DIM * DIM * sizeof(bf16);        // 0.5 MB
  const size_t bias_bytes = (size_t)NBLK * LEAF * (LEAF + 1) * 4;    // 17.3 MB
  bf16*  wq_t = (bf16*)ws;
  bf16*  wp_t = (bf16*)(ws + wq_bytes);
  float* bias = (float*)(ws + wq_bytes + wp_bytes);
  unsigned char* mask = (unsigned char*)(ws + wq_bytes + wp_bytes + bias_bytes);

  {
    int n = DIM * QKV_N + DIM * DIM;
    convert_weights_kernel<<<(n + 255) / 256, 256, 0, stream>>>(qkv_w, proj_w, wq_t, wp_t);
  }
  {
    long n = (long)NBLK * LEAF * (LEAF + 1);
    init_mask_bias_kernel<<<(int)((n + 255) / 256), 256, 0, stream>>>(bias, mask);
  }
  scatter_edges_kernel<<<(NEDGES + 255) / 256, 256, 0, stream>>>(
      ei, ei + NEDGES, ev, bias, mask);

  leaf_attn_kernel<<<NBLK, 256, LDS_BYTES, stream>>>(
      x, qkv_b, proj_b, eg_w, eg_b, wq_t, wp_t, bias, mask, out);
}
